// GATCrossAttentionPI_81235011437204
// MI455X (gfx1250) — matrix operand order repaired and checked by reading, not compiled, not run on hardware
//
#include <hip/hip_runtime.h>
#include <math.h>

typedef __attribute__((ext_vector_type(16))) __bf16          v16bf;
typedef __attribute__((ext_vector_type(8)))  float           v8f;
typedef __attribute__((ext_vector_type(8)))  unsigned short  v8us;

union FragB { v16bf v; v8us h[2]; };

static __device__ __forceinline__ unsigned short f2bf(float f){
  unsigned int u = __float_as_uint(f);
  unsigned int r = (u + 0x7FFFu + ((u >> 16) & 1u)) >> 16;   // round-to-nearest-even
  return (unsigned short)r;
}

static __device__ __forceinline__ float warp_reduce_sum(float v){
  #pragma unroll
  for (int off = 16; off > 0; off >>= 1) v += __shfl_xor(v, off, 32);
  return v;
}

static __device__ __forceinline__ void atomicMaxF(float* addr, float val){
  int* ia = (int*)addr;
  int cur = __float_as_int(*addr);
  while (val > __int_as_float(cur)){
    int assumed = cur;
    cur = atomicCAS(ia, assumed, __float_as_int(val));
    if (cur == assumed) break;
  }
}

// ---------------------------------------------------------------- small utils
__global__ void k_fill(float* __restrict__ p, float v, long long n){
  long long i = (long long)blockIdx.x * blockDim.x + threadIdx.x;
  if (i < n) p[i] = v;
}

// fp32 -> bf16 elementwise
__global__ void k_cvt(const float* __restrict__ in, unsigned short* __restrict__ out, long long n){
  long long i = (long long)blockIdx.x * blockDim.x + threadIdx.x;
  if (i < n) out[i] = f2bf(in[i]);
}

// W[K,Nout] fp32 -> Wt[Npad,Kpad] bf16 transposed + zero-padded
__global__ void k_wt(const float* __restrict__ W, unsigned short* __restrict__ Wt,
                     int K, int Nout, int Kpad, long long total){
  long long i = (long long)blockIdx.x * blockDim.x + threadIdx.x;
  if (i >= total) return;
  int n = (int)(i / Kpad), k = (int)(i % Kpad);
  float v = (k < K && n < Nout) ? W[(long long)k * Nout + n] : 0.f;
  Wt[i] = f2bf(v);
}

// metal_features[B,8] fp32 -> [B,32] bf16 zero-padded
__global__ void k_metal_pad(const float* __restrict__ metal, unsigned short* __restrict__ out, int total){
  int i = blockIdx.x * blockDim.x + threadIdx.x;
  if (i >= total) return;
  int b = i >> 5, j = i & 31;
  out[i] = f2bf(j < 8 ? metal[b * 8 + j] : 0.f);
}

// ---------------------------------------------------------------- WMMA GEMM
// Y[M,Nout] = act(Xbf[M,Kpad] @ Wt[Npad,Kpad]^T + bias)
// bf16 inputs (pre-converted, padded), f32 accumulate. No guards except final store.
// NT = 16-wide output tiles per wave (A fragment reused NT times).
template<int NT>
__global__ void k_linear_t(const unsigned short* __restrict__ Xbf,
                           const unsigned short* __restrict__ Wt,
                           const float* __restrict__ bias,
                           float* __restrict__ Y, unsigned short* __restrict__ Ybf,
                           int M, int Kpad, int Nout, int Npad, int act)
{
  const int lane    = threadIdx.x & 31;
  const int wave    = (blockIdx.x * blockDim.x + threadIdx.x) >> 5;
  const int groupsN = Npad / (16 * NT);
  const int tilesM  = M >> 4;
  if (wave >= tilesM * groupsN) return;            // wave-uniform
  const int rowBase = (wave / groupsN) << 4;
  const int colBase = (wave % groupsN) * (16 * NT);
  const int r  = lane & 15;
  const int hi = lane >> 4;

  v8f zero = {};
  v8f acc[NT];
  #pragma unroll
  for (int t = 0; t < NT; ++t) acc[t] = zero;

  // A: lane row = rowBase+r; K runs [base, base+15], base = hi*16
  const v8us* ap = (const v8us*)(Xbf + (size_t)(rowBase + r) * Kpad + (hi << 4));
  // B (transposed weights): lane col; K runs [base, base+15], base = hi*16
  const v8us* bp[NT];
  #pragma unroll
  for (int t = 0; t < NT; ++t)
    bp[t] = (const v8us*)(Wt + (size_t)(colBase + t * 16 + r) * Kpad + (hi << 4));

  for (int kc = 0; kc < Kpad; kc += 32){
    FragB a;
    a.h[0] = ap[0];                 // K base+0..7   -> VGPR0-3
    a.h[1] = ap[1];                 // K base+8..15  -> VGPR4-7
    ap += 4;                        // advance 32 K
    #pragma unroll
    for (int t = 0; t < NT; ++t){
      FragB b;
      b.h[0] = bp[t][0];            // K base+0..7
      b.h[1] = bp[t][1];            // K base+8..15
      bp[t] += 4;
      acc[t] = __builtin_amdgcn_wmma_f32_16x16x32_bf16(false, a.v, false, b.v,
                                                       (short)0, acc[t], false, false);
    }
  }

  const int orowBase = rowBase + (hi << 3);
  #pragma unroll
  for (int t = 0; t < NT; ++t){
    int col = colBase + t * 16 + r;
    if (col < Nout){
      float bb = bias ? bias[col] : 0.f;
      #pragma unroll
      for (int j = 0; j < 8; ++j){
        float v = acc[t][j] + bb;
        if (act == 1)      v = fmaxf(v, 0.f);
        else if (act == 2) v = (v > 0.f) ? v : (__expf(v) - 1.f);
        size_t idx = (size_t)(orowBase + j) * Nout + col;
        if (Y)   Y[idx]   = v;
        if (Ybf) Ybf[idx] = f2bf(v);
      }
    }
  }
}

// ---------------------------------------------------------------- GAT edge phase
__global__ void k_node_dots(const float* __restrict__ hw, const float* __restrict__ as_,
                            const float* __restrict__ ad_, float* __restrict__ es,
                            float* __restrict__ ed, int n){
  int wave = (blockIdx.x * blockDim.x + threadIdx.x) >> 5;
  int lane = threadIdx.x & 31;
  if (wave >= n) return;
  const float* hr = hw + (long long)wave * 128;
  float ps = 0.f, pd = 0.f;
  #pragma unroll
  for (int j = 0; j < 4; ++j){
    int c = lane + j * 32;
    float hv = hr[c];
    ps += hv * as_[c];
    pd += hv * ad_[c];
  }
  ps = warp_reduce_sum(ps);
  pd = warp_reduce_sum(pd);
  if (lane == 0){ es[wave] = ps; ed[wave] = pd; }
}

__global__ void k_edge_max(const int* __restrict__ ei, long long E,
                           const float* __restrict__ es, const float* __restrict__ ed,
                           float* __restrict__ m){
  long long e = (long long)blockIdx.x * blockDim.x + threadIdx.x;
  if (e >= E) return;
  int s = ei[e], d = ei[E + e];
  float x = es[s] + ed[d];
  x = (x > 0.f) ? x : 0.2f * x;
  atomicMaxF(&m[d], x);
}

__global__ void k_edge_accum(const int* __restrict__ ei, long long E,
                             const float* __restrict__ es, const float* __restrict__ ed,
                             const float* __restrict__ m, const float* __restrict__ hw,
                             float* __restrict__ z, float* __restrict__ acc){
  long long wave = ((long long)blockIdx.x * blockDim.x + threadIdx.x) >> 5;
  int lane = threadIdx.x & 31;
  if (wave >= E) return;
  int s = ei[wave], d = ei[E + wave];
  float x = es[s] + ed[d];
  x = (x > 0.f) ? x : 0.2f * x;
  float w = __expf(x - m[d]);
  if (lane == 0) atomicAdd(&z[d], w);
  const float* hs = hw + (long long)s * 128;
  float* ad = acc + (long long)d * 128;
  #pragma unroll
  for (int j = 0; j < 4; ++j){
    int c = lane + j * 32;
    atomicAdd(&ad[c], w * hs[c]);
  }
}

// P = ELU(acc / (z+eps)); also emit bf16 copy for the next GEMM
__global__ void k_finish_elu(float* __restrict__ P, unsigned short* __restrict__ Pbf,
                             const float* __restrict__ z, long long total){
  long long i = (long long)blockIdx.x * blockDim.x + threadIdx.x;
  if (i >= total) return;
  float v = P[i] / (z[i >> 7] + 1e-16f);
  v = (v > 0.f) ? v : (__expf(v) - 1.f);
  P[i]   = v;
  Pbf[i] = f2bf(v);
}

// ---------------------------------------------------------------- pooling
__global__ void k_pool(const float* __restrict__ h, const int* __restrict__ batch,
                       float* __restrict__ pooled, float* __restrict__ cnt, int n){
  int wave = (blockIdx.x * blockDim.x + threadIdx.x) >> 5;
  int lane = threadIdx.x & 31;
  if (wave >= n) return;
  int b = batch[wave];
  const float* hr = h + (long long)wave * 128;
  float* pr = pooled + (long long)b * 128;
  if (lane == 0) atomicAdd(&cnt[b], 1.f);
  #pragma unroll
  for (int j = 0; j < 4; ++j){
    int c = lane + j * 32;
    atomicAdd(&pr[c], hr[c]);
  }
}

__global__ void k_pool_div(float* __restrict__ pooled, unsigned short* __restrict__ pbf,
                           const float* __restrict__ cnt, int total){
  int i = blockIdx.x * blockDim.x + threadIdx.x;
  if (i >= total) return;
  float v = pooled[i] / fmaxf(cnt[i >> 7], 1.f);
  pooled[i] = v;
  pbf[i] = f2bf(v);
}

// ---------------------------------------------------------------- cross attention
__global__ void k_scores(const float* __restrict__ kbuf, const float* __restrict__ q,
                         const int* __restrict__ batch, float* __restrict__ scores,
                         float* __restrict__ sm, int n){
  int wave = (blockIdx.x * blockDim.x + threadIdx.x) >> 5;
  int lane = threadIdx.x & 31;
  if (wave >= n) return;
  int b = batch[wave];
  const float* kr = kbuf + (long long)wave * 128;
  const float* qr = q + (long long)b * 128;
  float p = 0.f;
  #pragma unroll
  for (int j = 0; j < 4; ++j){ int c = lane + j * 32; p += kr[c] * qr[c]; }
  p = warp_reduce_sum(p);
  if (lane == 0){
    float sc = p * 0.08838834764831845f;   // 1/sqrt(128)
    scores[wave] = sc;
    atomicMaxF(&sm[b], sc);
  }
}

__global__ void k_attn_accum(const float* __restrict__ vbuf, const float* __restrict__ scores,
                             const float* __restrict__ sm, const int* __restrict__ batch,
                             float* __restrict__ sz, float* __restrict__ attn, int n){
  int wave = (blockIdx.x * blockDim.x + threadIdx.x) >> 5;
  int lane = threadIdx.x & 31;
  if (wave >= n) return;
  int b = batch[wave];
  float w = __expf(scores[wave] - sm[b]);
  if (lane == 0) atomicAdd(&sz[b], w);
  const float* vr = vbuf + (long long)wave * 128;
  float* ar = attn + (long long)b * 128;
  #pragma unroll
  for (int j = 0; j < 4; ++j){
    int c = lane + j * 32;
    atomicAdd(&ar[c], w * vr[c]);
  }
}

__global__ void k_attn_div(float* __restrict__ attn, const float* __restrict__ sz, int total){
  int i = blockIdx.x * blockDim.x + threadIdx.x;
  if (i < total) attn[i] /= (sz[i >> 7] + 1e-16f);
}

// combined bf16 [B,288] = [pi_out(10) | attn_out(128) | metal_embed(128) | zero-pad(22)]
__global__ void k_concat(const float* __restrict__ pi, const float* __restrict__ attn,
                         const float* __restrict__ me, unsigned short* __restrict__ comb, int total){
  int i = blockIdx.x * blockDim.x + threadIdx.x;
  if (i >= total) return;
  int b = i / 288, j = i - b * 288;
  float v = 0.f;
  if (j < 10)       v = pi[b * 10 + j];
  else if (j < 138) v = attn[b * 128 + (j - 10)];
  else if (j < 266) v = me[b * 128 + (j - 138)];
  comb[i] = f2bf(v);
}

// ---------------------------------------------------------------- host helpers
static inline void fill(float* p, float v, long long n, hipStream_t s){
  k_fill<<<(int)((n + 255) / 256), 256, 0, s>>>(p, v, n);
}
static inline void launch_linear(const unsigned short* Xbf, const unsigned short* Wt,
                                 const float* bias, float* Y, unsigned short* Ybf,
                                 int M, int Kpad, int Nout, int act, hipStream_t s){
  int Npad = (Nout + 15) & ~15;
  if (Npad % 64 == 0){
    long long waves = (long long)(M >> 4) * (Npad / 64);
    k_linear_t<4><<<(int)((waves * 32 + 255) / 256), 256, 0, s>>>(Xbf, Wt, bias, Y, Ybf, M, Kpad, Nout, Npad, act);
  } else {
    long long waves = (long long)(M >> 4) * (Npad / 16);
    k_linear_t<1><<<(int)((waves * 32 + 255) / 256), 256, 0, s>>>(Xbf, Wt, bias, Y, Ybf, M, Kpad, Nout, Npad, act);
  }
}

extern "C" void kernel_launch(void* const* d_in, const int* in_sizes, int n_in,
                              void* d_out, int out_size, void* d_ws, size_t ws_size,
                              hipStream_t stream) {
  (void)in_sizes; (void)n_in; (void)out_size; (void)ws_size;
  const int  N = 131072, B = 2048, H = 128;
  const long long E = 1048576;

  const float* x     = (const float*)d_in[0];
  const float* metal = (const float*)d_in[1];
  const int*   ei    = (const int*)d_in[2];
  const int*   batch = (const int*)d_in[3];
  const float* W0    = (const float*)d_in[4];
  const float* Wr    = (const float*)d_in[5];
  const float* gb    = (const float*)d_in[6];
  const float* a_src = (const float*)d_in[7];
  const float* a_dst = (const float*)d_in[8];
  const float* mW0 = (const float*)d_in[9],  *mb0 = (const float*)d_in[10];
  const float* mW1 = (const float*)d_in[11], *mb1 = (const float*)d_in[12];
  const float* Wq = (const float*)d_in[13], *bq = (const float*)d_in[14];
  const float* Wk = (const float*)d_in[15], *bk = (const float*)d_in[16];
  const float* Wv = (const float*)d_in[17], *bv = (const float*)d_in[18];
  const float* piW[5], *piB[5], *rW[5], *rB[5];
  for (int i = 0; i < 5; ++i){
    piW[i] = (const float*)d_in[19 + 2 * i];
    piB[i] = (const float*)d_in[20 + 2 * i];
    rW[i]  = (const float*)d_in[29 + 2 * i];
    rB[i]  = (const float*)d_in[30 + 2 * i];
  }

  // ---------------- workspace: fp32 region then bf16 region
  float* w = (float*)d_ws;
  long long o = 0;
  float* hw     = w + o; o += (long long)N * H;
  float* P      = w + o; o += (long long)N * H;
  float* es     = w + o; o += N;                 // reused for attention scores
  float* ed     = w + o; o += N;
  float* m      = w + o; o += N;
  float* z      = w + o; o += N;
  float* pooled = w + o; o += (long long)B * H;
  float* cnt    = w + o; o += B;
  float* me     = w + o; o += (long long)B * H;
  float* qb     = w + o; o += (long long)B * H;
  float* sm     = w + o; o += B;
  float* sz     = w + o; o += B;
  float* attn   = w + o; o += (long long)B * H;

  unsigned short* ub = (unsigned short*)(w + o);
  long long u = 0;
  unsigned short* x_bf      = ub + u; u += (long long)N * 64;
  unsigned short* P_bf      = ub + u; u += (long long)N * H;
  unsigned short* pooled_bf = ub + u; u += (long long)B * H;
  unsigned short* t0_bf     = ub + u; u += (long long)B * 512;
  unsigned short* t1_bf     = ub + u; u += (long long)B * 512;
  unsigned short* me_bf     = ub + u; u += (long long)B * H;
  unsigned short* metal_bf  = ub + u; u += (long long)B * 32;
  unsigned short* comb_bf   = ub + u; u += (long long)B * 288;

  // transposed+padded bf16 weights: [Npad, Kpad]
  struct WD { const float* W; int K, Nout, Kpad, Npad; unsigned short* dst; };
  WD wd[19] = {
    { W0,                 64, 128,  64, 128, nullptr },
    { Wr + 0 * H * H,    128, 128, 128, 128, nullptr },
    { Wr + 1 * H * H,    128, 128, 128, 128, nullptr },
    { Wr + 2 * H * H,    128, 128, 128, 128, nullptr },
    { mW0,                 8, 128,  32, 128, nullptr },
    { mW1,               128, 128, 128, 128, nullptr },
    { Wq,                128, 128, 128, 128, nullptr },
    { Wk,                128, 128, 128, 128, nullptr },
    { Wv,                128, 128, 128, 128, nullptr },
    { piW[0],            128, 512, 128, 512, nullptr },
    { piW[1],            512, 512, 512, 512, nullptr },
    { piW[2],            512, 512, 512, 512, nullptr },
    { piW[3],            512, 256, 512, 256, nullptr },
    { piW[4],            256,  10, 256,  16, nullptr },
    { rW[0],             266, 512, 288, 512, nullptr },
    { rW[1],             512, 512, 512, 512, nullptr },
    { rW[2],             512, 512, 512, 512, nullptr },
    { rW[3],             512, 256, 512, 256, nullptr },
    { rW[4],             256,   1, 256,  16, nullptr },
  };
  for (int i = 0; i < 19; ++i){
    wd[i].dst = ub + u;
    u += (long long)wd[i].Npad * wd[i].Kpad;
  }
  for (int i = 0; i < 19; ++i){
    long long tot = (long long)wd[i].Npad * wd[i].Kpad;
    k_wt<<<(int)((tot + 255) / 256), 256, 0, stream>>>(wd[i].W, wd[i].dst,
                                                       wd[i].K, wd[i].Nout, wd[i].Kpad, tot);
  }

  const int nodeWaveBlocks = (N * 32) / 256;       // 16384
  const long long NH = (long long)N * H;

  // convert x to bf16 once
  k_cvt<<<(int)(((long long)N * 64 + 255) / 256), 256, 0, stream>>>(x, x_bf, (long long)N * 64);

  // ---- GAT stack (4 layers)
  const unsigned short* hin_bf = x_bf; int Kin = 64;
  for (int layer = 0; layer < 4; ++layer){
    launch_linear(hin_bf, wd[layer].dst, gb + layer * H, hw, nullptr, N, Kin, H, 0, stream);
    k_node_dots<<<nodeWaveBlocks, 256, 0, stream>>>(hw, a_src + layer * H, a_dst + layer * H, es, ed, N);
    fill(m, -3.0e38f, N, stream);
    fill(z, 0.f, N, stream);
    fill(P, 0.f, NH, stream);
    k_edge_max<<<(int)((E + 255) / 256), 256, 0, stream>>>(ei, E, es, ed, m);
    k_edge_accum<<<(int)((E * 32 + 255) / 256), 256, 0, stream>>>(ei, E, es, ed, m, hw, z, P);
    k_finish_elu<<<(int)((NH + 255) / 256), 256, 0, stream>>>(P, P_bf, z, NH);
    hin_bf = P_bf; Kin = H;
  }

  // ---- global mean pool
  fill(pooled, 0.f, (long long)B * H, stream);
  fill(cnt, 0.f, B, stream);
  k_pool<<<nodeWaveBlocks, 256, 0, stream>>>(P, batch, pooled, cnt, N);
  k_pool_div<<<(B * H + 255) / 256, 256, 0, stream>>>(pooled, pooled_bf, cnt, B * H);

  // ---- pi head: 128->512->512->512->256->10 (final into d_out[B..))
  float* outF  = (float*)d_out;
  float* piOut = outF + B;                         // d_out = [energy(B) | pi_out(B*10)]
  launch_linear(pooled_bf, wd[9].dst,  piB[0], nullptr, t0_bf, B, 128, 512, 1, stream);
  launch_linear(t0_bf,     wd[10].dst, piB[1], nullptr, t1_bf, B, 512, 512, 1, stream);
  launch_linear(t1_bf,     wd[11].dst, piB[2], nullptr, t0_bf, B, 512, 512, 1, stream);
  launch_linear(t0_bf,     wd[12].dst, piB[3], nullptr, t1_bf, B, 512, 256, 1, stream);
  launch_linear(t1_bf,     wd[13].dst, piB[4], piOut,  nullptr, B, 256, 10, 0, stream);

  // ---- metal embedding: 8->128->128 (relu), then q projection
  k_metal_pad<<<(B * 32 + 255) / 256, 256, 0, stream>>>(metal, metal_bf, B * 32);
  launch_linear(metal_bf, wd[4].dst, mb0, nullptr, t0_bf, B, 32, 128, 1, stream);   // t0_bf as [B,128]
  launch_linear(t0_bf,    wd[5].dst, mb1, me, me_bf,     B, 128, 128, 1, stream);
  launch_linear(me_bf,    wd[6].dst, bq,  qb, nullptr,   B, 128, 128, 0, stream);

  // ---- cross attention over nodes
  launch_linear(P_bf, wd[7].dst, bk, hw, nullptr, N, 128, 128, 0, stream);  // k into hw
  fill(sm, -3.0e38f, B, stream);
  fill(sz, 0.f, B, stream);
  fill(attn, 0.f, (long long)B * H, stream);
  k_scores<<<nodeWaveBlocks, 256, 0, stream>>>(hw, qb, batch, es, sm, N);
  launch_linear(P_bf, wd[8].dst, bv, hw, nullptr, N, 128, 128, 0, stream);  // v overwrites hw
  k_attn_accum<<<nodeWaveBlocks, 256, 0, stream>>>(hw, es, sm, batch, sz, attn, N);
  k_attn_div<<<(B * H + 255) / 256, 256, 0, stream>>>(attn, sz, B * H);

  // ---- concat (bf16, padded to 288) + regressor: 266->512->512->512->256->1
  k_concat<<<(B * 288 + 255) / 256, 256, 0, stream>>>(piOut, attn, me, comb_bf, B * 288);
  launch_linear(comb_bf, wd[14].dst, rB[0], nullptr, t0_bf, B, 288, 512, 1, stream);
  launch_linear(t0_bf,   wd[15].dst, rB[1], nullptr, t1_bf, B, 512, 512, 1, stream);
  launch_linear(t1_bf,   wd[16].dst, rB[2], nullptr, t0_bf, B, 512, 512, 1, stream);
  launch_linear(t0_bf,   wd[17].dst, rB[3], nullptr, t1_bf, B, 512, 256, 1, stream);
  launch_linear(t1_bf,   wd[18].dst, rB[4], outF,  nullptr, B, 256, 1, 0, stream);
}
